// Transformer_17995912970858
// MI455X (gfx1250) — matrix operand order repaired and checked by reading, not compiled, not run on hardware
//
#include <hip/hip_runtime.h>

// ---------------------------------------------------------------------------
// Types for CDNA5 WMMA (wave32)
// ---------------------------------------------------------------------------
typedef __bf16 bf16_t;
typedef bf16_t v16bf __attribute__((ext_vector_type(16)));
typedef float  v8f   __attribute__((ext_vector_type(8)));
typedef float  v4f   __attribute__((ext_vector_type(4)));

#define WMMA_BF16(a, b, c) \
  __builtin_amdgcn_wmma_f32_16x16x32_bf16(false, (a), false, (b), (short)0, (c), false, false)

// Model constants
#define TOKS   4096      // B*N
#define NSEQ   2048
#define DIMM   1024
#define DCOND  4096
#define FFIN   2730
#define FFPAD  2752      // 2730 padded to multiple of 32
#define FF2    5460
#define NEGBIG (-1e30f)

// ---------------------------------------------------------------------------
// Fragment loaders
// A 16x32 bf16 fragment: lane row = lane&15, kb = 16*(lane>=16):
//   elements 0..15 = K kb..kb+15
// p must point at &A[row*lda + kchunk + kb] (16B aligned)
// ---------------------------------------------------------------------------
static __device__ __forceinline__ v16bf load_a_frag(const float* p) {
  const v4f* q = (const v4f*)p;
  v4f a0 = q[0], a1 = q[1], a2 = q[2], a3 = q[3];
  v16bf r;
  r[0]  = (bf16_t)a0[0]; r[1]  = (bf16_t)a0[1]; r[2]  = (bf16_t)a0[2]; r[3]  = (bf16_t)a0[3];
  r[4]  = (bf16_t)a1[0]; r[5]  = (bf16_t)a1[1]; r[6]  = (bf16_t)a1[2]; r[7]  = (bf16_t)a1[3];
  r[8]  = (bf16_t)a2[0]; r[9]  = (bf16_t)a2[1]; r[10] = (bf16_t)a2[2]; r[11] = (bf16_t)a2[3];
  r[12] = (bf16_t)a3[0]; r[13] = (bf16_t)a3[1]; r[14] = (bf16_t)a3[2]; r[15] = (bf16_t)a3[3];
  return r;
}

// B 32x16 bf16 fragment from 16 contiguous floats (K contiguous in memory).
// lane col = lane&15, element e = K (16*(lane>=16) + e). p: 16B aligned.
static __device__ __forceinline__ v16bf load_b_contig(const float* p) {
  const v4f* q = (const v4f*)p;
  v4f b0 = q[0], b1 = q[1], b2 = q[2], b3 = q[3];
  v16bf r;
  r[0]  = (bf16_t)b0[0]; r[1]  = (bf16_t)b0[1]; r[2]  = (bf16_t)b0[2]; r[3]  = (bf16_t)b0[3];
  r[4]  = (bf16_t)b1[0]; r[5]  = (bf16_t)b1[1]; r[6]  = (bf16_t)b1[2]; r[7]  = (bf16_t)b1[3];
  r[8]  = (bf16_t)b2[0]; r[9]  = (bf16_t)b2[1]; r[10] = (bf16_t)b2[2]; r[11] = (bf16_t)b2[3];
  r[12] = (bf16_t)b3[0]; r[13] = (bf16_t)b3[1]; r[14] = (bf16_t)b3[2]; r[15] = (bf16_t)b3[3];
  return r;
}

// ---------------------------------------------------------------------------
// Generic WMMA GEMM: C[M,N] = A[M,Kpad] * B[Kb,N] (+ bias[n]).
// LDB is compile-time so strided B reads become immediate offsets off one base.
// A rows must be valid (zero-padded) out to Kpad (Kpad % 32 == 0, lda % 4 == 0).
// Wave tile: 32x64 (2 A-frags x 4 B-frags, 8 WMMAs per 32-K chunk).
// Block: 8 waves = 256 rows. Guarded path only for N-edge blocks / K tail.
// ---------------------------------------------------------------------------
template <int LDB>
__global__ __launch_bounds__(256) void gemm_bf16_kernel(
    const float* __restrict__ A, int lda,
    const float* __restrict__ B,
    const float* __restrict__ bias,
    float* __restrict__ C, int ldc,
    int M, int N, int Kpad, int Kb)
{
  const int wid  = threadIdx.x >> 5;
  const int lane = threadIdx.x & 31;
  const int half = lane >> 4;
  const int l16  = lane & 15;
  const int m0 = (blockIdx.y * 8 + wid) * 32;
  const int n0 = blockIdx.x * 64;
  if (m0 >= M) return;

  const float* arow0 = A + (size_t)(m0 + l16) * lda + half * 16;
  const float* arow1 = arow0 + (size_t)16 * lda;
  const int bk = half * 16;
  v8f acc[2][4] = {};

  const bool interior = (n0 + 64 <= N);
  const int Kfull = interior ? (Kb & ~31) : 0;
  int k = 0;

  if (interior) {
    // Single per-lane base pointer; all 64 B loads use immediate offsets.
    const float* bp = B + (size_t)bk * LDB + n0 + l16;
    for (; k < Kfull; k += 32) {
      v16bf a0 = load_a_frag(arow0 + k);
      v16bf a1 = load_a_frag(arow1 + k);
#pragma unroll
      for (int t = 0; t < 4; ++t) {
        v16bf bf;
#pragma unroll
        for (int e = 0; e < 16; ++e) bf[e] = (bf16_t)bp[(size_t)e * LDB + t * 16];
        acc[0][t] = WMMA_BF16(a0, bf, acc[0][t]);
        acc[1][t] = WMMA_BF16(a1, bf, acc[1][t]);
      }
      bp += (size_t)32 * LDB;
    }
  }
  // Guarded remainder: K tail (Kb % 32 != 0) and/or N-edge blocks.
  for (; k < Kpad; k += 32) {
    v16bf a0 = load_a_frag(arow0 + k);
    v16bf a1 = load_a_frag(arow1 + k);
    int kk = k + bk;
#pragma unroll
    for (int t = 0; t < 4; ++t) {
      int col = n0 + t * 16 + l16;
      v16bf bf;
#pragma unroll
      for (int e = 0; e < 16; ++e) {
        int kkk = kk + e;
        float v = (col < N && kkk < Kb) ? B[(size_t)kkk * LDB + col] : 0.0f;
        bf[e] = (bf16_t)v;
      }
      acc[0][t] = WMMA_BF16(a0, bf, acc[0][t]);
      acc[1][t] = WMMA_BF16(a1, bf, acc[1][t]);
    }
  }

#pragma unroll
  for (int rr = 0; rr < 2; ++rr)
#pragma unroll
    for (int t = 0; t < 4; ++t) {
      int col = n0 + t * 16 + l16;
      if (col >= N) continue;
      float bv = bias ? bias[col] : 0.0f;
#pragma unroll
      for (int r = 0; r < 8; ++r) {
        int row = m0 + rr * 16 + r + half * 8;
        C[(size_t)row * ldc + col] = acc[rr][t][r] + bv;
      }
    }
}

// ---------------------------------------------------------------------------
// Block reduction (256 threads = 8 wave32)
// ---------------------------------------------------------------------------
static __device__ __forceinline__ float block_reduce(float v, float* sh) {
#pragma unroll
  for (int d = 16; d > 0; d >>= 1) v += __shfl_xor(v, d, 32);
  __syncthreads();
  if ((threadIdx.x & 31) == 0) sh[threadIdx.x >> 5] = v;
  __syncthreads();
  float r = 0.0f;
#pragma unroll
  for (int i = 0; i < 8; ++i) r += sh[i];
  return r;
}

static __device__ __forceinline__ void mod_spans(const int* mp, int b,
                                                 int& o0, int& e0, int& o1, int& e1) {
  o0 = mp[b * 6 + 1]; e0 = o0 + mp[b * 6 + 2];
  o1 = mp[b * 6 + 4]; e1 = o1 + mp[b * 6 + 5];
}

// ---------------------------------------------------------------------------
// Fourier features F[tok, 1024]: cols 0..511 sin, 512..1023 cos
// ---------------------------------------------------------------------------
__global__ __launch_bounds__(256) void fourier_kernel(
    const float* __restrict__ times, const float* __restrict__ fw, float* __restrict__ F)
{
  int idx = blockIdx.x * 256 + threadIdx.x;
  int tok = idx >> 10, kk = idx & 1023;
  float t = times[tok] * 6.283185307179586f;
  float v = (kk < 512) ? sinf(t * fw[kk]) : cosf(t * fw[kk - 512]);
  F[idx] = v;
}

// cond = silu(P + times*time_w[0,:])  (bias already added by GEMM), in place
__global__ __launch_bounds__(256) void cond_silu_kernel(
    float* __restrict__ COND, const float* __restrict__ times, const float* __restrict__ time_w0)
{
  int idx = blockIdx.x * 256 + threadIdx.x;
  int tok = idx >> 12, c = idx & 4095;
  float v = COND[idx] + times[tok] * time_w0[c];
  COND[idx] = v / (1.0f + expf(-v));
}

// ---------------------------------------------------------------------------
// LayerNorm -> FiLM/ln_gamma -> RMSNorm(rms_gamma). One block per token.
// ---------------------------------------------------------------------------
__global__ __launch_bounds__(256) void adaln_kernel(
    const float* __restrict__ x, const float* __restrict__ GB,
    const float* __restrict__ ln_g, const float* __restrict__ rms_g,
    const int* __restrict__ mp, float* __restrict__ h2)
{
  __shared__ float sh[8];
  int tok = blockIdx.x, tid = threadIdx.x;
  const float* xr = x + (size_t)tok * DIMM;
  float xv[4], s = 0.0f, ss = 0.0f;
#pragma unroll
  for (int i = 0; i < 4; ++i) { xv[i] = xr[tid + 256 * i]; s += xv[i]; ss += xv[i] * xv[i]; }
  s  = block_reduce(s, sh);
  ss = block_reduce(ss, sh);
  float mu = s * (1.0f / DIMM);
  float var = ss * (1.0f / DIMM) - mu * mu;
  float rstd = rsqrtf(var + 1e-5f);
  int b = tok >> 11, n = tok & 2047, o0, e0, o1, e1;
  mod_spans(mp, b, o0, e0, o1, e1);
  bool any = (n >= o0 && n < e0) || (n >= o1 && n < e1);
  float hv[4], s2 = 0.0f;
#pragma unroll
  for (int i = 0; i < 4; ++i) {
    int c = tid + 256 * i;
    float h = (xv[i] - mu) * rstd;
    if (any) h = h * (GB[(size_t)tok * 2048 + c] + 1.0f) + GB[(size_t)tok * 2048 + 1024 + c];
    else     h = h * (ln_g[c] + 1.0f);
    hv[i] = h; s2 += h * h;
  }
  s2 = block_reduce(s2, sh);
  float inv = 32.0f / fmaxf(sqrtf(s2), 1e-12f);   // sqrt(DIM)=32
#pragma unroll
  for (int i = 0; i < 4; ++i) {
    int c = tid + 256 * i;
    h2[(size_t)tok * DIMM + c] = hv[i] * inv * (rms_g[c] + 1.0f);
  }
}

// ---------------------------------------------------------------------------
// Split QKV + rotary (interleaved pairs) + fold q*SCALE. One thread per pair.
// Outputs [B,H,N,64].
// ---------------------------------------------------------------------------
__global__ __launch_bounds__(256) void rotary_split_kernel(
    const float* __restrict__ QKV, const float* __restrict__ rot,
    float* __restrict__ qr, float* __restrict__ kr, float* __restrict__ vb)
{
  int idx = blockIdx.x * 256 + threadIdx.x;     // tok*512 + h*32 + p
  int p = idx & 31, h = (idx >> 5) & 15, tok = idx >> 9;
  int b = tok >> 11, n = tok & 2047, d0 = 2 * p;
  size_t qo = (size_t)tok * 3072 + h * 64 + d0;
  float q0 = QKV[qo],        q1 = QKV[qo + 1];
  float k0 = QKV[qo + 1024], k1 = QKV[qo + 1025];
  float v0 = QKV[qo + 2048], v1 = QKV[qo + 2049];
  float r0 = rot[(size_t)n * 64 + d0], r1 = rot[(size_t)n * 64 + d0 + 1];
  float c0 = cosf(r0), s0 = sinf(r0), c1 = cosf(r1), s1 = sinf(r1);
  size_t oo = ((size_t)(b * 16 + h) * NSEQ + n) * 64 + d0;
  qr[oo]     = (q0 * c0 - q1 * s0) * 0.125f;    // SCALE = 64^-0.5
  qr[oo + 1] = (q1 * c1 + q0 * s1) * 0.125f;
  kr[oo]     =  k0 * c0 - k1 * s0;
  kr[oo + 1] =  k1 * c1 + k0 * s1;
  vb[oo]     =  v0;
  vb[oo + 1] =  v1;
}

// ---------------------------------------------------------------------------
// Flash attention with softcap + causal|modality mask. One wave per
// (b, h, 16-row query tile). out[tok, h*64+d].
// ---------------------------------------------------------------------------
__global__ __launch_bounds__(256) void attn_kernel(
    const float* __restrict__ qr, const float* __restrict__ kr,
    const float* __restrict__ vbuf, float* __restrict__ out,
    const int* __restrict__ mp)
{
  __shared__ bf16_t plds[8][512];               // per-wave 16x32 bf16 P tile
  const int wid = threadIdx.x >> 5, lane = threadIdx.x & 31;
  const int half = lane >> 4, l16 = lane & 15;
  const int gw = blockIdx.x * 8 + wid;          // 0..4095
  const int qt = gw & 127, h = (gw >> 7) & 15, b = gw >> 11;
  const int bh = b * 16 + h, i0 = qt * 16;
  const float* qb  = qr   + (size_t)bh * NSEQ * 64;
  const float* kp0 = kr   + (size_t)bh * NSEQ * 64;
  const float* vp0 = vbuf + (size_t)bh * NSEQ * 64;
  int o0, e0, o1, e1; mod_spans(mp, b, o0, e0, o1, e1);

  v16bf qa0 = load_a_frag(qb + (size_t)(i0 + l16) * 64 + 0  + half * 16);
  v16bf qa1 = load_a_frag(qb + (size_t)(i0 + l16) * 64 + 32 + half * 16);

  float mrow[8], lrow[8], alpha[8];
  v8f oacc[4] = {};
#pragma unroll
  for (int r = 0; r < 8; ++r) { mrow[r] = NEGBIG; lrow[r] = 0.0f; }
  bf16_t* pw = &plds[wid][0];

  for (int j0 = 0; j0 < NSEQ; j0 += 32) {
    v8f s0 = {}, s1 = {};
    {
      const float* kp = kp0 + (size_t)(j0 + l16) * 64 + half * 16;
      s0 = WMMA_BF16(qa0, load_b_contig(kp), s0);
      s0 = WMMA_BF16(qa1, load_b_contig(kp + 32), s0);
      kp = kp0 + (size_t)(j0 + 16 + l16) * 64 + half * 16;
      s1 = WMMA_BF16(qa0, load_b_contig(kp), s1);
      s1 = WMMA_BF16(qa1, load_b_contig(kp + 32), s1);
    }
#pragma unroll
    for (int r = 0; r < 8; ++r) {
      int i = i0 + r + half * 8;
      int ja = j0 + l16, jb = j0 + 16 + l16;
      float sa = tanhf(s0[r] * 0.02f) * 50.0f;  // softcap 50
      float sb = tanhf(s1[r] * 0.02f) * 50.0f;
      bool aa = (i >= ja) || (i >= o0 && ja < e0) || (i >= o1 && ja < e1);
      bool ab = (i >= jb) || (i >= o0 && jb < e0) || (i >= o1 && jb < e1);
      if (!aa) sa = NEGBIG;
      if (!ab) sb = NEGBIG;
      float mx = fmaxf(sa, sb);
#pragma unroll
      for (int d = 1; d < 16; d <<= 1) mx = fmaxf(mx, __shfl_xor(mx, d, 32));
      float mn = fmaxf(mrow[r], mx);
      float al = __expf(mrow[r] - mn);
      float pa = __expf(sa - mn), pb = __expf(sb - mn);
      float rs = pa + pb;
#pragma unroll
      for (int d = 1; d < 16; d <<= 1) rs += __shfl_xor(rs, d, 32);
      lrow[r] = lrow[r] * al + rs;
      mrow[r] = mn;
      alpha[r] = al;
      int row = r + half * 8;
      pw[row * 32 + l16]      = (bf16_t)pa;
      pw[row * 32 + 16 + l16] = (bf16_t)pb;
    }
    asm volatile("s_wait_dscnt 0" ::: "memory");
    v16bf pf;
    {
      const bf16_t* pr = pw + l16 * 32 + half * 16;
#pragma unroll
      for (int e = 0; e < 8; ++e) pf[e] = pr[e];
#pragma unroll
      for (int e = 0; e < 8; ++e) pf[8 + e] = pr[8 + e];
    }
#pragma unroll
    for (int t = 0; t < 4; ++t) {
      const float* vp = vp0 + (size_t)(j0 + half * 16) * 64 + t * 16 + l16;
      v16bf vf;
#pragma unroll
      for (int e = 0; e < 16; ++e) vf[e] = (bf16_t)vp[(size_t)e * 64];
      v8f c = oacc[t];
#pragma unroll
      for (int r = 0; r < 8; ++r) c[r] *= alpha[r];
      oacc[t] = WMMA_BF16(pf, vf, c);
    }
    asm volatile("s_wait_dscnt 0" ::: "memory");
  }
#pragma unroll
  for (int t = 0; t < 4; ++t)
#pragma unroll
    for (int r = 0; r < 8; ++r) {
      int i = i0 + r + half * 8;
      out[((size_t)b * NSEQ + i) * DIMM + h * 64 + t * 16 + l16] = oacc[t][r] / lrow[r];
    }
}

// ---------------------------------------------------------------------------
// Gated residual: x += any ? proj*sigmoid(GT) : proj*(ls+1)
// ---------------------------------------------------------------------------
__global__ __launch_bounds__(256) void gate_add_kernel(
    float* __restrict__ x, const float* __restrict__ proj,
    const float* __restrict__ GT, const float* __restrict__ ls,
    const int* __restrict__ mp)
{
  int idx = blockIdx.x * 256 + threadIdx.x;
  int tok = idx >> 10, c = idx & 1023;
  int b = tok >> 11, n = tok & 2047, o0, e0, o1, e1;
  mod_spans(mp, b, o0, e0, o1, e1);
  bool any = (n >= o0 && n < e0) || (n >= o1 && n < e1);
  float pv = proj[idx];
  float g = any ? pv / (1.0f + expf(-GT[idx])) : pv * (ls[c] + 1.0f);
  x[idx] += g;
}

// GEGLU: GG[tok, 0..2751] = (c<2730) ? u * gelu_exact(g) : 0  (K-padded)
__global__ __launch_bounds__(256) void geglu_kernel(
    const float* __restrict__ H1, float* __restrict__ GG)
{
  int idx = blockIdx.x * 256 + threadIdx.x;
  int tok = idx / FFPAD, c = idx - tok * FFPAD;
  float v = 0.0f;
  if (c < FFIN) {
    float u = H1[(size_t)tok * FF2 + c];
    float g = H1[(size_t)tok * FF2 + FFIN + c];
    v = u * 0.5f * g * (1.0f + erff(g * 0.70710678118654752f));
  }
  GG[idx] = v;
}

// Final RMSNorm -> d_out
__global__ __launch_bounds__(256) void final_rms_kernel(
    const float* __restrict__ x, const float* __restrict__ gm, float* __restrict__ out)
{
  __shared__ float sh[8];
  int tok = blockIdx.x, tid = threadIdx.x;
  const float* xr = x + (size_t)tok * DIMM;
  float xv[4], ss = 0.0f;
#pragma unroll
  for (int i = 0; i < 4; ++i) { xv[i] = xr[tid + 256 * i]; ss += xv[i] * xv[i]; }
  ss = block_reduce(ss, sh);
  float inv = 32.0f / fmaxf(sqrtf(ss), 1e-12f);
#pragma unroll
  for (int i = 0; i < 4; ++i) {
    int c = tid + 256 * i;
    out[(size_t)tok * DIMM + c] = xv[i] * inv * (gm[c] + 1.0f);
  }
}

// ---------------------------------------------------------------------------
// Host orchestration
// ---------------------------------------------------------------------------
static inline void launch_gemm(int ldb, hipStream_t stream,
                               const float* A, int lda, const float* B,
                               const float* bias, float* C, int ldc,
                               int M, int N, int Kpad, int Kb)
{
  dim3 grid((N + 63) / 64, (M + 255) / 256), blk(256);
  switch (ldb) {
    case 1024: gemm_bf16_kernel<1024><<<grid, blk, 0, stream>>>(A, lda, B, bias, C, ldc, M, N, Kpad, Kb); break;
    case 2048: gemm_bf16_kernel<2048><<<grid, blk, 0, stream>>>(A, lda, B, bias, C, ldc, M, N, Kpad, Kb); break;
    case 3072: gemm_bf16_kernel<3072><<<grid, blk, 0, stream>>>(A, lda, B, bias, C, ldc, M, N, Kpad, Kb); break;
    case 4096: gemm_bf16_kernel<4096><<<grid, blk, 0, stream>>>(A, lda, B, bias, C, ldc, M, N, Kpad, Kb); break;
    case 5460: gemm_bf16_kernel<5460><<<grid, blk, 0, stream>>>(A, lda, B, bias, C, ldc, M, N, Kpad, Kb); break;
    default: break;
  }
}

extern "C" void kernel_launch(void* const* d_in, const int* in_sizes, int n_in,
                              void* d_out, int out_size, void* d_ws, size_t ws_size,
                              hipStream_t stream)
{
  (void)in_sizes; (void)n_in; (void)out_size; (void)ws_size;
  const float* x      = (const float*)d_in[0];
  const float* times  = (const float*)d_in[1];
  const float* rot    = (const float*)d_in[2];
  const float* fw     = (const float*)d_in[3];
  const float* time_w = (const float*)d_in[4];
  const float* time_b = (const float*)d_in[5];
  const float* afw    = (const float*)d_in[6];
  const float* afb    = (const float*)d_in[7];
  const float* azw    = (const float*)d_in[8];
  const float* azb    = (const float*)d_in[9];
  const float* alng   = (const float*)d_in[10];
  const float* als    = (const float*)d_in[11];
  const float* armsg  = (const float*)d_in[12];
  const float* wqkv   = (const float*)d_in[13];
  const float* wout   = (const float*)d_in[14];
  const float* ffw    = (const float*)d_in[15];
  const float* ffb    = (const float*)d_in[16];
  const float* fzw    = (const float*)d_in[17];
  const float* fzb    = (const float*)d_in[18];
  const float* flng   = (const float*)d_in[19];
  const float* fls    = (const float*)d_in[20];
  const float* frmsg  = (const float*)d_in[21];
  const float* w1     = (const float*)d_in[22];
  const float* b1     = (const float*)d_in[23];
  const float* w2     = (const float*)d_in[24];
  const float* b2     = (const float*)d_in[25];
  const float* fing   = (const float*)d_in[26];
  const int*   mpos   = (const int*)d_in[27];
  float* outp = (float*)d_out;

  // Workspace layout (floats), with aliasing of dead buffers
  float* wsf  = (float*)d_ws;
  float* COND = wsf;                 // 4096*4096
  float* GB   = wsf + 16777216;      // 4096*2048 (also F at start)
  float* F    = GB;                  // 4096*1024
  float* GT   = wsf + 25165824;      // 4096*1024
  float* H2   = wsf + 29360128;      // 4096*1024
  float* ATT  = wsf + 33554432;      // 4096*1024
  float* BIG  = wsf + 37748736;      // max(QKV 4096*3072, H1 4096*5460)
  float* SPL  = wsf + 60112896;      // qr/kr/vb (3*4096*1024) | GG (4096*2752)
  float* PROJ = wsf + 72695808;      // 4096*1024
  float* XC   = wsf + 76890112;      // 4096*1024
  float* QR = SPL, *KR = SPL + 4194304, *VB = SPL + 8388608, *GG = SPL;

  dim3 blk(256);
  hipMemcpyAsync(XC, x, (size_t)TOKS * DIMM * sizeof(float),
                 hipMemcpyDeviceToDevice, stream);

  // cond = silu(fourier @ time_w + time_b)
  fourier_kernel<<<16384, blk, 0, stream>>>(times, fw, F);
  launch_gemm(4096, stream, F, 1024, time_w + DCOND, time_b, COND, DCOND,
              TOKS, DCOND, 1024, 1024);
  cond_silu_kernel<<<65536, blk, 0, stream>>>(COND, times, time_w);

  for (int l = 0; l < 2; ++l) {
    const float* Lafw = afw + (size_t)l * DCOND * 2048;
    const float* Lafb = afb + (size_t)l * 2048;
    const float* Lazw = azw + (size_t)l * DCOND * DIMM;
    const float* Lazb = azb + (size_t)l * DIMM;
    const float* Lwqkv = wqkv + (size_t)l * DIMM * 3072;
    const float* Lwout = wout + (size_t)l * DIMM * DIMM;
    const float* Lffw = ffw + (size_t)l * DCOND * 2048;
    const float* Lffb = ffb + (size_t)l * 2048;
    const float* Lfzw = fzw + (size_t)l * DCOND * DIMM;
    const float* Lfzb = fzb + (size_t)l * DIMM;
    const float* Lw1 = w1 + (size_t)l * DIMM * FF2;
    const float* Lb1 = b1 + (size_t)l * FF2;
    const float* Lw2 = w2 + (size_t)l * FFIN * DIMM;
    const float* Lb2 = b2 + (size_t)l * DIMM;

    // ---- attention branch ----
    launch_gemm(2048, stream, COND, DCOND, Lafw, Lafb, GB, 2048, TOKS, 2048, DCOND, DCOND);
    launch_gemm(1024, stream, COND, DCOND, Lazw, Lazb, GT, DIMM, TOKS, DIMM, DCOND, DCOND);
    adaln_kernel<<<TOKS, blk, 0, stream>>>(XC, GB, alng + (size_t)l * DIMM,
                                           armsg + (size_t)l * DIMM, mpos, H2);
    launch_gemm(3072, stream, H2, DIMM, Lwqkv, nullptr, BIG, 3072, TOKS, 3072, DIMM, DIMM);
    rotary_split_kernel<<<8192, blk, 0, stream>>>(BIG, rot, QR, KR, VB);
    attn_kernel<<<512, blk, 0, stream>>>(QR, KR, VB, ATT, mpos);
    launch_gemm(1024, stream, ATT, DIMM, Lwout, nullptr, PROJ, DIMM, TOKS, DIMM, DIMM, DIMM);
    gate_add_kernel<<<16384, blk, 0, stream>>>(XC, PROJ, GT, als + (size_t)l * DIMM, mpos);

    // ---- feed-forward branch ----
    launch_gemm(2048, stream, COND, DCOND, Lffw, Lffb, GB, 2048, TOKS, 2048, DCOND, DCOND);
    launch_gemm(1024, stream, COND, DCOND, Lfzw, Lfzb, GT, DIMM, TOKS, DIMM, DCOND, DCOND);
    adaln_kernel<<<TOKS, blk, 0, stream>>>(XC, GB, flng + (size_t)l * DIMM,
                                           frmsg + (size_t)l * DIMM, mpos, H2);
    launch_gemm(5460, stream, H2, DIMM, Lw1, Lb1, BIG, FF2, TOKS, FF2, DIMM, DIMM);
    geglu_kernel<<<44032, blk, 0, stream>>>(BIG, GG);
    launch_gemm(1024, stream, GG, FFPAD, Lw2, Lb2, PROJ, DIMM, TOKS, DIMM, FFPAD, FFIN);
    gate_add_kernel<<<16384, blk, 0, stream>>>(XC, PROJ, GT, fls + (size_t)l * DIMM, mpos);
  }

  final_rms_kernel<<<TOKS, blk, 0, stream>>>(XC, fing, outp);
}
